// MultiHeadAttention_42511586296095
// MI455X (gfx1250) — matrix operand order repaired and checked by reading, not compiled, not run on hardware
//
#include <hip/hip_runtime.h>
#include <hip/hip_bf16.h>

// ---------------------------------------------------------------------------
// MHA with per-head full-width projections, bf16 WMMA pipeline for gfx1250.
//  x:[4,2048,768]  Wq/Wk/Wv:[12,768,768]  bq/bk/bv:[12,768]
//  W0:[9216,768]   b0:[768]   out:[4,2048,768] f32
//
// Pipeline: cvt x -> bf16 | transpose-cvt weights -> bf16 [N][K] |
//           qkv GEMMs (v written transposed) | q.k^T GEMM | row softmax |
//           attn.v GEMM | out-proj GEMM.
// All GEMMs: v_wmma_f32_16x16x32_bf16, 128x128x64 tiles, B always [N][K].
// Tile staging: TDM tensor_load_to_lds (double-buffered, s_wait_tensorcnt)
// when the builtin exists; vector global->LDS copy otherwise.
// Workspace requirement ~1.08 GB.
// ---------------------------------------------------------------------------

typedef unsigned short u16;
typedef __attribute__((ext_vector_type(16))) __bf16 v16bf;
typedef __attribute__((ext_vector_type(8)))  float  v8f;
typedef __attribute__((ext_vector_type(4)))  unsigned int u32x4;

#if __has_builtin(__builtin_amdgcn_tensor_load_to_lds) && \
    __has_builtin(__builtin_amdgcn_s_wait_tensorcnt)
#define MHA_HAS_TDM 1
#endif

// The tensor_load_to_lds builtin is not modeled as writing LDS.  Countermeasures
// history: a bare "memory" clobber is ignored (ldsbuf doesn't escape, rounds
// 2-3: loads folded to undef); volatile loads work but lower to serialized
// flat_load SCOPE_SYS (round 4).  What we use now:
//  (a) escape the LDS pointer INTO an asm with a memory clobber, and
//  (b) a phantom store to ldsbuf inside the K-loop, predicated on a kernel
//      argument condition that is never true at runtime, so MemorySSA sees a
//      potential in-loop write and cannot fold/hoist the plain ds loads.
__device__ __forceinline__ void lds_written_by_tdm(const void* p) {
  asm volatile("" : : "v"(p) : "memory");
}

__device__ __forceinline__ u16 f32_to_bf16(float f) {
  unsigned int u = __float_as_uint(f);
  u += 0x7FFFu + ((u >> 16) & 1u);         // round-to-nearest-even
  return (u16)(u >> 16);
}
__device__ __forceinline__ float bf16_to_f32(u16 h) {
  return __uint_as_float(((unsigned int)h) << 16);
}

// ---------------------------------------------------------------------------
__global__ void cvt_f32_bf16(const float* __restrict__ in, u16* __restrict__ out, long n) {
  long i = (long)blockIdx.x * blockDim.x + threadIdx.x;
  long stride = (long)gridDim.x * blockDim.x;
  for (; i < n; i += stride) out[i] = f32_to_bf16(in[i]);
}

// Transpose-convert one [R][C] f32 slice per blockIdx.z into [C][R] bf16.
__global__ __launch_bounds__(256) void tcvt_f32_bf16(const float* __restrict__ in,
                                                     u16* __restrict__ out,
                                                     int R, int C) {
  __shared__ float t[32][33];
  const float* src = in + (size_t)blockIdx.z * R * C;
  u16* dst = out + (size_t)blockIdx.z * R * C;
  const int r0 = blockIdx.y * 32, c0 = blockIdx.x * 32;
  const int tx = threadIdx.x & 31, ty = threadIdx.x >> 5;   // 8 rows / pass
  #pragma unroll
  for (int i = 0; i < 4; ++i)
    t[ty + 8 * i][tx] = src[(size_t)(r0 + ty + 8 * i) * C + c0 + tx];
  __syncthreads();
  #pragma unroll
  for (int i = 0; i < 4; ++i)
    dst[(size_t)(c0 + ty + 8 * i) * R + r0 + tx] = f32_to_bf16(t[tx][ty + 8 * i]);
}

// ---------------------------------------------------------------------------
// TDM descriptor helpers (CDNA5 ISA ch.8: D# groups).
// ---------------------------------------------------------------------------
#ifdef MHA_HAS_TDM
typedef __attribute__((ext_vector_type(4))) unsigned int tdm_u4;
typedef __attribute__((ext_vector_type(4))) int          tdm_i4;
typedef __attribute__((ext_vector_type(8))) int          tdm_i8;

__device__ __forceinline__ void tdm_setbits(unsigned int* w, int lo, int nbits,
                                            unsigned long long val) {
  for (int i = 0; i < nbits; ) {
    int word = (lo + i) >> 5;
    int sh   = (lo + i) & 31;
    int take = 32 - sh; if (take > nbits - i) take = nbits - i;
    unsigned long long mask = (take >= 64) ? ~0ull : ((1ull << take) - 1ull);
    w[word] |= (unsigned int)(((val >> i) & mask) << sh);
    i += take;
  }
}

// Issue a 2D TDM load: tile_rows x tile_k bf16 elements from gptr (row pitch
// ld elements) into LDS byte offset lds_off, packed rows of 128B followed by
// 16B of padding (pad_interval=32 DW, pad_amount=4 DW) -> 144B LDS rows,
// matching the compute side's LDS_LD = 72 ushorts.
__device__ __forceinline__ void tdm_load_tile(const u16* gptr, unsigned int lds_off,
                                              int tile_k, int tile_rows, int ld) {
  unsigned int g0w[4] = {0, 0, 0, 0};
  unsigned int g1w[8] = {0, 0, 0, 0, 0, 0, 0, 0};
  unsigned long long ga = (unsigned long long)(size_t)gptr;

  g0w[0] = 1u;                                        // count=1, is_restore=0
  g0w[1] = lds_off;                                   // lds_addr (bytes)
  g0w[2] = (unsigned int)(ga & 0xFFFFFFFFull);        // global_addr[31:0]
  g0w[3] = (unsigned int)((ga >> 32) & 0x01FFFFFFull) // global_addr[56:32]
         | (2u << 30);                                // type=2 ("image")

  tdm_setbits(g1w, 16, 2, 1);                         // data_size = 2 bytes
  tdm_setbits(g1w, 20, 1, 1);                         // pad_enable
  tdm_setbits(g1w, 22, 3, 4);                         // pad_interval: 32 DWORDs
  tdm_setbits(g1w, 25, 7, 3);                         // pad_amount: 4 DWORDs
  tdm_setbits(g1w, 48, 32, (unsigned long long)tile_k);    // tensor_dim0
  tdm_setbits(g1w, 80, 32, (unsigned long long)tile_rows); // tensor_dim1
  tdm_setbits(g1w, 112, 16, (unsigned long long)tile_k);   // tile_dim0
  tdm_setbits(g1w, 128, 16, (unsigned long long)tile_rows);// tile_dim1
  tdm_setbits(g1w, 160, 48, (unsigned long long)ld);       // tensor_dim0_stride

  tdm_u4 g0; tdm_i8 g1;
  #pragma unroll
  for (int i = 0; i < 4; ++i) g0[i] = g0w[i];
  #pragma unroll
  for (int i = 0; i < 8; ++i) g1[i] = (int)g1w[i];
  tdm_i4 z4 = (tdm_i4)0;
#if defined(__clang_major__) && __clang_major__ >= 23
  tdm_i8 z8 = (tdm_i8)0;
  __builtin_amdgcn_tensor_load_to_lds(g0, g1, z4, z4, z8, 0);
#else
  __builtin_amdgcn_tensor_load_to_lds(g0, g1, z4, z4, 0);
#endif
}
#endif // MHA_HAS_TDM

// ---------------------------------------------------------------------------
// GEMM:  C = alpha * A(MxK) * B^T + bias.  B memory is ALWAYS [N][K] row-major.
// C written row-major (transC=0, idx=row*ldc+col) or transposed (transC=1,
// idx=col*ldc+row).  Batched over blockIdx.z = outer*nInner + inner.
// M,N multiples of 128; K multiple of 64 (true for all shapes here).
// ---------------------------------------------------------------------------
#define BLK_M 128
#define BLK_N 128
#define BLK_K 64
#define LDS_LD (BLK_K + 8)              // 72 ushorts = 144B rows (16B aligned)
#define TILE_B (BLK_M * LDS_LD * 2)     // 18432 bytes per tile

union Frag { v16bf v; u32x4 q[2]; };

__global__ __launch_bounds__(256) void wmma_gemm_bf16(
    const u16* __restrict__ A, const u16* __restrict__ B,
    const float* __restrict__ bias,
    u16* __restrict__ Cb, float* __restrict__ Cf,
    int M, int N, int K, int lda, int ldb, int ldc,
    long sAo, long sAi, long sBo, long sBi, long sCo, long sCi,
    long sbo, long sbi, int nInner, float alpha, int transC)
{
  // [buffer][A/B][tile]; TDM descriptors assume this sits at LDS offset 0.
  __shared__ __align__(16) u16 ldsbuf[2][2][BLK_M * LDS_LD];

  const int tid   = threadIdx.x;
  const int lane  = tid & 31;           // wave32
  const int wv    = tid >> 5;           // 8 waves
  const int half  = lane >> 4;
  const int l16   = lane & 15;
  const int waveM = wv >> 1;            // 0..3 -> 32 rows
  const int waveN = wv & 1;             // 0..1 -> 64 cols

  const int outer = blockIdx.z / nInner;
  const int inner = blockIdx.z % nInner;
  A += (size_t)outer * sAo + (size_t)inner * sAi;
  B += (size_t)outer * sBo + (size_t)inner * sBi;
  if (bias) bias += (size_t)outer * sbo + (size_t)inner * sbi;
  const size_t cOff = (size_t)outer * sCo + (size_t)inner * sCi;

  const int rowBase = blockIdx.y * BLK_M;
  const int colBase = blockIdx.x * BLK_N;
  (void)M; (void)N;

  v8f acc[2][4];
  #pragma unroll
  for (int mi = 0; mi < 2; ++mi)
    #pragma unroll
    for (int ni = 0; ni < 4; ++ni)
      acc[mi][ni] = (v8f)(0.0f);

#ifdef MHA_HAS_TDM
  // Prologue: wave 0 issues the first K-panel into buffer 0.
  if (wv == 0) {
    tdm_load_tile(A + (size_t)rowBase * lda, 0 * TILE_B, BLK_K, BLK_M, lda);
    tdm_load_tile(B + (size_t)colBase * ldb, 1 * TILE_B, BLK_K, BLK_M, ldb);
  }
  lds_written_by_tdm(&ldsbuf[0][0][0]);
  int cur = 0;
#endif

  for (int k0 = 0; k0 < K; k0 += BLK_K) {
#ifdef MHA_HAS_TDM
    __syncthreads();                    // all waves done reading buffer cur^1
    const int haveNext = (k0 + BLK_K) < K;
    if (wv == 0) {
      if (haveNext) {                   // issue next panel into the other buffer
        const int nb = cur ^ 1;
        tdm_load_tile(A + (size_t)rowBase * lda + k0 + BLK_K,
                      (unsigned)(nb * 2 + 0) * TILE_B, BLK_K, BLK_M, lda);
        tdm_load_tile(B + (size_t)colBase * ldb + k0 + BLK_K,
                      (unsigned)(nb * 2 + 1) * TILE_B, BLK_K, BLK_M, ldb);
        __builtin_amdgcn_s_wait_tensorcnt(2);   // TDM in-order: current pair done
      } else {
        __builtin_amdgcn_s_wait_tensorcnt(0);
      }
    }
    __syncthreads();                    // release all waves onto buffer cur
    // LDS changed behind the compiler's back: escape the pointer + clobber,
    // plus a phantom store (never executes: nInner >= 1 always) that gives
    // MemorySSA an in-loop MayDef on ldsbuf so the ds loads below can be
    // neither undef-folded nor hoisted out of the loop.
    lds_written_by_tdm(&ldsbuf[cur][0][0]);
    if (nInner < 0) ldsbuf[0][0][tid] = (u16)K;
    u16 (*As)[LDS_LD] = (u16(*)[LDS_LD])ldsbuf[cur][0];
    u16 (*Bs)[LDS_LD] = (u16(*)[LDS_LD])ldsbuf[cur][1];
#else
    u16 (*As)[LDS_LD] = (u16(*)[LDS_LD])ldsbuf[0][0];
    u16 (*Bs)[LDS_LD] = (u16(*)[LDS_LD])ldsbuf[0][1];
    __syncthreads();
    // vector staging: both tiles are [128 rows][64 k] row-major copies
    #pragma unroll
    for (int i = 0; i < 4; ++i) {
      int u = tid + 256 * i;            // 1024 uint4 per tile
      int r = u >> 3;
      int c = (u & 7) << 3;
      u32x4 da = *(const u32x4*)(A + (size_t)(rowBase + r) * lda + k0 + c);
      *(u32x4*)(&As[r][c]) = da;
      u32x4 db = *(const u32x4*)(B + (size_t)(colBase + r) * ldb + k0 + c);
      *(u32x4*)(&Bs[r][c]) = db;
    }
    if (k0 + BLK_K < K) {
      __builtin_prefetch(A + (size_t)(rowBase + (tid >> 1)) * lda + k0 + BLK_K, 0, 1);
      __builtin_prefetch(B + (size_t)(colBase + (tid >> 1)) * ldb + k0 + BLK_K, 0, 1);
    }
    __syncthreads();
#endif

    // ---- compute: two 16x16x32 wmma K-steps per staged panel ----
    #pragma unroll
    for (int ks = 0; ks < 2; ++ks) {
      Frag fa[2], fb[4];
      // A frag (ISA 7.12.2, 16-bit A 16x32): lanes 0-15 K 0..7,16..23;
      // lanes 16-31 K 8..15,24..31.
      #pragma unroll
      for (int mi = 0; mi < 2; ++mi) {
        const u16* p = &As[waveM * 32 + mi * 16 + l16][ks * 32 + half * 8];
        fa[mi].q[0] = *(const u32x4*)p;
        fa[mi].q[1] = *(const u32x4*)(p + 16);
      }
      // B frag (32x16): lanes 0-15 K 0..7,16..23; lanes 16-31 K 8..15,24..31.
      #pragma unroll
      for (int ni = 0; ni < 4; ++ni) {
        const u16* p = &Bs[waveN * 64 + ni * 16 + l16][ks * 32 + half * 8];
        fb[ni].q[0] = *(const u32x4*)p;
        fb[ni].q[1] = *(const u32x4*)(p + 16);
      }
      #pragma unroll
      for (int mi = 0; mi < 2; ++mi)
        #pragma unroll
        for (int ni = 0; ni < 4; ++ni)
          acc[mi][ni] = __builtin_amdgcn_wmma_f32_16x16x32_bf16(
              false, fa[mi].v, false, fb[ni].v, (short)0, acc[mi][ni], false, false);
    }
#ifdef MHA_HAS_TDM
    cur ^= 1;
#endif
  }

  // ---- epilogue: C/D layout -- VGPR r: M = half*8 + r, N = l16 ----
  #pragma unroll
  for (int ni = 0; ni < 4; ++ni) {
    int col = colBase + waveN * 64 + ni * 16 + l16;
    float bv = bias ? bias[col] : 0.0f;
    #pragma unroll
    for (int mi = 0; mi < 2; ++mi) {
      #pragma unroll
      for (int r = 0; r < 8; ++r) {
        int row = rowBase + waveM * 32 + mi * 16 + half * 8 + r;
        float val = acc[mi][ni][r] * alpha + bv;
        size_t idx = transC ? (cOff + (size_t)col * ldc + row)
                            : (cOff + (size_t)row * ldc + col);
        if (Cf) Cf[idx] = val;
        else    Cb[idx] = f32_to_bf16(val);
      }
    }
  }
}

// ---------------------------------------------------------------------------
// In-place row softmax over 2048 bf16 scores (f32 math). One block per row.
// ---------------------------------------------------------------------------
__global__ __launch_bounds__(256) void softmax_rows_bf16(u16* __restrict__ S, int n) {
  __shared__ float red[256];
  const int tid = threadIdx.x;
  u16* row = S + (size_t)blockIdx.x * n;

  float v[8];
  float mx = -3.0e38f;
  #pragma unroll
  for (int i = 0; i < 8; ++i) {
    v[i] = bf16_to_f32(row[tid + 256 * i]);
    mx = fmaxf(mx, v[i]);
  }
  red[tid] = mx; __syncthreads();
  for (int s = 128; s > 0; s >>= 1) {
    if (tid < s) red[tid] = fmaxf(red[tid], red[tid + s]);
    __syncthreads();
  }
  mx = red[0]; __syncthreads();

  float sum = 0.0f;
  #pragma unroll
  for (int i = 0; i < 8; ++i) { v[i] = __expf(v[i] - mx); sum += v[i]; }
  red[tid] = sum; __syncthreads();
  for (int s = 128; s > 0; s >>= 1) {
    if (tid < s) red[tid] += red[tid + s];
    __syncthreads();
  }
  float inv = 1.0f / red[0];
  #pragma unroll
  for (int i = 0; i < 8; ++i) row[tid + 256 * i] = f32_to_bf16(v[i] * inv);
}

// ---------------------------------------------------------------------------
extern "C" void kernel_launch(void* const* d_in, const int* in_sizes, int n_in,
                              void* d_out, int out_size, void* d_ws, size_t ws_size,
                              hipStream_t stream) {
  (void)in_sizes; (void)n_in; (void)out_size; (void)ws_size;

  const int Bb = 4, S = 2048, E = 768, H = 12;
  const int BS = Bb * S;        // 8192
  const int HE = H * E;         // 9216

  const float* x  = (const float*)d_in[0];
  const float* Wq = (const float*)d_in[1];
  const float* Wk = (const float*)d_in[2];
  const float* Wv = (const float*)d_in[3];
  const float* bq = (const float*)d_in[4];
  const float* bk = (const float*)d_in[5];
  const float* bv = (const float*)d_in[6];
  const float* W0 = (const float*)d_in[7];
  const float* b0 = (const float*)d_in[8];
  float* out = (float*)d_out;

  // ---- workspace layout (bf16 buffers), total ~1.08 GB ----
  char* ws = (char*)d_ws;
  size_t off = 0;
  auto alloc = [&](size_t bytes) -> void* {
    off = (off + 255) & ~(size_t)255;
    void* p = ws + off; off += bytes; return p;
  };
  const size_t n_x = (size_t)BS * E;         // x bf16
  const size_t n_w = (size_t)H * E * E;      // per weight tensor (also W0)
  const size_t n_q = (size_t)H * BS * E;     // q/k: [h][b*s][e]; vT: [h][e][b*s]
  const size_t n_S = (size_t)Bb * H * S * S; // scores/probs [b][h][s][t]
  const size_t n_z = (size_t)BS * HE;        // z: [b*s][h*e]

  u16* xb  = (u16*)alloc(n_x * 2);
  u16* wqT = (u16*)alloc(n_w * 2);   // [h][E_out][E_in]
  u16* wkT = (u16*)alloc(n_w * 2);
  u16* wvT = (u16*)alloc(n_w * 2);
  u16* w0T = (u16*)alloc(n_w * 2);   // [E_out=768][HE=9216]
  u16* qb  = (u16*)alloc(n_q * 2);
  u16* kb  = (u16*)alloc(n_q * 2);
  u16* vT  = (u16*)alloc(n_q * 2);
  u16* Sb  = (u16*)alloc(n_S * 2);
  u16* zb  = (u16*)alloc(n_z * 2);

  dim3 blk(256);

  // ---- 1) convert / transpose-convert inputs to bf16 ----
  cvt_f32_bf16<<<1024, blk, 0, stream>>>(x, xb, (long)n_x);
  {
    dim3 gt(E / 32, E / 32, H);               // per-head [768][768] transpose
    tcvt_f32_bf16<<<gt, blk, 0, stream>>>(Wq, wqT, E, E);
    tcvt_f32_bf16<<<gt, blk, 0, stream>>>(Wk, wkT, E, E);
    tcvt_f32_bf16<<<gt, blk, 0, stream>>>(Wv, wvT, E, E);
    dim3 g0(E / 32, HE / 32, 1);              // [9216][768] -> [768][9216]
    tcvt_f32_bf16<<<g0, blk, 0, stream>>>(W0, w0T, HE, E);
  }

  // ---- 2) q/k/v projections: q[h] = x @ Wq[h] + bq[h] (B = WqT [N][K]) ----
  {
    dim3 g(E / BLK_N, BS / BLK_M, H);         // (6,64,12)
    const long sBi = (long)E * E, sbi = E;
    wmma_gemm_bf16<<<g, blk, 0, stream>>>(xb, wqT, bq, qb, nullptr,
        BS, E, E, E, E, E, 0, 0, 0, sBi, 0, (long)BS * E, 0, sbi, H, 1.0f, 0);
    wmma_gemm_bf16<<<g, blk, 0, stream>>>(xb, wkT, bk, kb, nullptr,
        BS, E, E, E, E, E, 0, 0, 0, sBi, 0, (long)BS * E, 0, sbi, H, 1.0f, 0);
    // v is written TRANSPOSED: vT[h][e][b*s], ldc = BS
    wmma_gemm_bf16<<<g, blk, 0, stream>>>(xb, wvT, bv, vT, nullptr,
        BS, E, E, E, E, BS, 0, 0, 0, sBi, 0, (long)E * BS, 0, sbi, H, 1.0f, 1);
  }

  // ---- 3) scores[b,h] = (q[b,h] @ k[b,h]^T) / sqrt(64); k is [N][K] already ----
  {
    dim3 g(S / BLK_N, S / BLK_M, Bb * H);     // (16,16,48), outer=b inner=h
    wmma_gemm_bf16<<<g, blk, 0, stream>>>(qb, kb, nullptr, Sb, nullptr,
        S, S, E, E, E, S,
        (long)S * E, (long)BS * E,            // A (q): b, h
        (long)S * E, (long)BS * E,            // B (k): b, h
        (long)H * S * S, (long)S * S,         // C (scores): b, h
        0, 0, H, 0.125f, 0);
  }

  // ---- 4) softmax over key axis, in place ----
  softmax_rows_bf16<<<Bb * H * S, blk, 0, stream>>>(Sb, S);

  // ---- 5) y[b,h] = P[b,h] @ v[b,h]; B = vT slice [E][S], ldb = BS ----
  {
    dim3 g(E / BLK_N, S / BLK_M, Bb * H);     // (6,16,48)
    wmma_gemm_bf16<<<g, blk, 0, stream>>>(Sb, vT, nullptr, zb, nullptr,
        S, E, S, S, BS, HE,
        (long)H * S * S, (long)S * S,         // A (P): b, h
        (long)S, (long)E * BS,                // B (vT): b -> col offset, h
        (long)S * HE, (long)E,                // C (z): b, h -> col block h*E
        0, 0, H, 1.0f, 0);
  }

  // ---- 6) out = z @ W0 + b0 (f32 out); B = w0T [768][9216] ----
  {
    dim3 g(E / BLK_N, BS / BLK_M, 1);         // (6,64,1)
    wmma_gemm_bf16<<<g, blk, 0, stream>>>(zb, w0T, b0, nullptr, out,
        BS, E, HE, HE, HE, E,
        0, 0, 0, 0, 0, 0, 0, 0, 1, 1.0f, 0);
  }
}
